// LSTMNet_static_3753801417255
// MI455X (gfx1250) — matrix operand order repaired and checked by reading, not compiled, not run on hardware
//
#include <hip/hip_runtime.h>
#include <hip/hip_bf16.h>
#include <math.h>

// ---------------------------------------------------------------------------
// LSTMNet on gfx1250 (MI455X): WMMA f16 GEMMs + persistent recurrent kernel
// with async-to-LDS staging and software-pipelined LDS->WMMA feeding.
// ---------------------------------------------------------------------------

typedef __attribute__((ext_vector_type(16))) _Float16 v16h;
typedef __attribute__((ext_vector_type(8)))  float    v8f;

constexpr int Bc  = 32;            // batch
constexpr int Tc  = 1024;          // sequence length
constexpr int Ic  = 128;           // input dim
constexpr int Hc  = 512;           // hidden dim
constexpr int G4c = 4 * Hc;        // 2048 gate rows
constexpr int Oc  = 128;           // output dim
constexpr int Mc  = Bc * Tc;       // 32768 rows of the big GEMMs
constexpr int NWG = 32;            // workgroups in persistent recurrent kernel

union HVec {
  v16h  v;
  uint4 q[2];
};

__device__ __forceinline__ float sigmoidf_(float x) {
  return 1.0f / (1.0f + __expf(-x));
}

// gfx1250 async global->LDS copy (ASYNCcnt-tracked, no VGPR round trip).
__device__ __forceinline__ void async_load_lds_b128(unsigned lds_off,
                                                    const void* gptr) {
  asm volatile("global_load_async_to_lds_b128 %0, %1, off"
               :: "v"(lds_off), "v"(gptr)
               : "memory");
}
__device__ __forceinline__ void wait_asynccnt0() {
  asm volatile("s_wait_asynccnt 0x0" ::: "memory");
}
__device__ __forceinline__ void wait_asynccnt1() {
  asm volatile("s_wait_asynccnt 0x1" ::: "memory");
}

// ---------------------------------------------------------------------------
// f32 -> f16 conversion / bias combine
// ---------------------------------------------------------------------------
__global__ void f32_to_f16_kernel(const float* __restrict__ in,
                                  _Float16* __restrict__ out, int n) {
  int i = blockIdx.x * blockDim.x + threadIdx.x;
  if (i < n) out[i] = (_Float16)in[i];
}

__global__ void bias_sum_kernel(const float* __restrict__ a,
                                const float* __restrict__ b,
                                float* __restrict__ o, int n) {
  int i = blockIdx.x * blockDim.x + threadIdx.x;
  if (i < n) o[i] = a[i] + b[i];
}

// ---------------------------------------------------------------------------
// Generic WMMA GEMM:  C[M,N] = A[M,K] @ Bw[N,K]^T + bias[N]
// A, Bw row-major f16. 256 threads = 8 waves; WG tile 64x64; each wave owns a
// 16(M) x 32(N) sub-tile (two 16x16 accumulators). K multiple of 32.
// Double-buffered fragment loads; sched_group_barrier pins loads-ahead-of-
// WMMA so VMEM latency overlaps the matrix pipe.
// ---------------------------------------------------------------------------
template <typename OutT>
__global__ void __launch_bounds__(256)
wmma_gemm_kernel(const _Float16* __restrict__ A,
                 const _Float16* __restrict__ Bw,
                 const float* __restrict__ bias,
                 OutT* __restrict__ C,
                 int Mdim, int Ndim, int Kdim) {
  const int lane  = threadIdx.x & 31;
  const int wave  = threadIdx.x >> 5;
  const bool hi   = lane >= 16;
  const int ln    = lane & 15;
  const int tileM = blockIdx.y * 64;
  const int tileN = blockIdx.x * 64;
  const int msub  = wave & 3;   // 4 M sub-tiles
  const int npair = wave >> 2;  // 0 or 1

  const int rowA = tileM + msub * 16 + ln;
  const _Float16* arow  = A  + (size_t)rowA * Kdim;
  const int rowB0 = tileN + npair * 32 + ln;
  const _Float16* brow0 = Bw + (size_t)rowB0 * Kdim;
  const _Float16* brow1 = brow0 + (size_t)16 * Kdim;

  const int abase = hi ? 8 : 0;    // A: K offset of this lane-half
  const int bbase = hi ? 8 : 0;    // B: K offset of this lane-half

  HVec a[2], b0[2], b1[2];
  auto load_frag = [&](int k, int s) {
    a[s].q[0]  = *(const uint4*)(arow  + k + abase);
    a[s].q[1]  = *(const uint4*)(arow  + k + abase + 16);
    b0[s].q[0] = *(const uint4*)(brow0 + k + bbase);
    b0[s].q[1] = *(const uint4*)(brow0 + k + bbase + 16);
    b1[s].q[0] = *(const uint4*)(brow1 + k + bbase);
    b1[s].q[1] = *(const uint4*)(brow1 + k + bbase + 16);
  };

  v8f acc0 = {};
  v8f acc1 = {};
  const int nk = Kdim >> 5;
  load_frag(0, 0);
  for (int i = 0; i < nk; ++i) {
    const int s = i & 1;
    if (i + 1 < nk) load_frag((i + 1) << 5, s ^ 1);
    acc0 = __builtin_amdgcn_wmma_f32_16x16x32_f16(false, a[s].v, false, b0[s].v,
                                                  (short)0, acc0, false, false);
    acc1 = __builtin_amdgcn_wmma_f32_16x16x32_f16(false, a[s].v, false, b1[s].v,
                                                  (short)0, acc1, false, false);
    // Desired per-iteration order: next-iter VMEM reads, then this iter's
    // two WMMAs (0x20 = VMEM read, 0x8 = MFMA/WMMA).
    __builtin_amdgcn_sched_group_barrier(0x020, 6, 0);
    __builtin_amdgcn_sched_group_barrier(0x008, 2, 0);
  }

  const int n0 = tileN + npair * 32 + ln;
  const int n1 = n0 + 16;
  const float bv0 = bias ? bias[n0] : 0.0f;
  const float bv1 = bias ? bias[n1] : 0.0f;
  const int mrow = tileM + msub * 16 + abase;  // D: M = r + (hi?8:0)
#pragma unroll
  for (int r = 0; r < 8; ++r) {
    C[(size_t)(mrow + r) * Ndim + n0] = (OutT)(acc0[r] + bv0);
    C[(size_t)(mrow + r) * Ndim + n1] = (OutT)(acc1[r] + bv1);
  }
}

// ---------------------------------------------------------------------------
// Persistent LSTM recurrence. 32 WGs x 256 threads (8 waves). WG owns 16
// hidden units -> 64 gate rows = 4 gates x 16. Wave w: gate = w>>1, m-tile =
// w&1. W_hh slice lives in VGPRs for the whole T loop (16 x v16h per wave).
// Per step: h_{t-1} and this step's xg slice staged via gfx1250 async-to-LDS
// loads; A fragments fed through a 4-deep LDS->VGPR pipeline interleaved
// with the WMMA chain. Steps separated by an agent-scope atomic grid barrier.
// ---------------------------------------------------------------------------
__global__ void __launch_bounds__(256, 1)
lstm_recurrent_kernel(const _Float16* __restrict__ xg,    // [B,T,4H] f16
                      const _Float16* __restrict__ Whh,   // [4H,H]   f16
                      _Float16* __restrict__ hseq,        // [B,T,H]  f16 out
                      unsigned* __restrict__ barrier,
                      unsigned phase_base) {
  __shared__ __align__(16) _Float16 sh_h[Bc][Hc];        // 32 KB
  __shared__ __align__(16) float    sh_gate[4][Bc][16];  // 8 KB
  __shared__ __align__(16) float    sh_c[Bc][16];        // 2 KB
  __shared__ __align__(16) _Float16 sh_xg[4][Bc][16];    // 4 KB

  const int tid   = threadIdx.x;
  const int lane  = tid & 31;
  const int wave  = tid >> 5;
  const bool hi   = lane >= 16;
  const int ln    = lane & 15;
  const int gate  = wave >> 1;   // 0..3 (i,f,g,o)
  const int mtile = wave & 1;    // batch half
  const int hs    = blockIdx.x * 16;  // hidden slice base

  // Wave-relative LDS byte addresses for async destinations.
  const unsigned ldsa_h  = (unsigned)(uintptr_t)(&sh_h[0][0]);
  const unsigned ldsa_xg = (unsigned)(uintptr_t)(&sh_xg[0][0][0]);

  // zero h (step 0 input) and c
  for (int i = tid; i < (Bc * Hc) / 8; i += 256)
    ((uint4*)sh_h)[i] = make_uint4(0u, 0u, 0u, 0u);
  for (int i = tid; i < Bc * 16; i += 256)
    ((float*)sh_c)[i] = 0.0f;

  // Preload W_hh fragments for this wave's (gate, n=hs..hs+15) rows.
  v16h bfrag[16];
  {
    const _Float16* wrow = Whh + (size_t)(gate * Hc + hs + ln) * Hc;
    const int bbase = hi ? 8 : 0;
#pragma unroll
    for (int kt = 0; kt < 16; ++kt) {
      HVec b;
      b.q[0] = *(const uint4*)(wrow + kt * 32 + bbase);
      b.q[1] = *(const uint4*)(wrow + kt * 32 + bbase + 16);
      bfrag[kt] = b.v;
    }
  }
  __syncthreads();

  // xg async staging geometry: 128 chunks of 32B; thread -> 16B half-chunk.
  const int xchunk = tid >> 1;        // 0..127 : gate*32 + b
  const int xgate  = xchunk >> 5;     // 0..3
  const int xb     = xchunk & 31;     // 0..31
  const int xhalf  = (tid & 1) * 8;   // 0 or 8 halfs
  const unsigned xg_dst =
      ldsa_xg + (unsigned)(((xgate * Bc + xb) * 16 + xhalf) * 2);

  const int abase = hi ? 8 : 0;
  const _Float16* hrow = &sh_h[mtile * 16 + ln][0];

  for (int t = 0; t < Tc; ++t) {
    if (t > 0) {
      // async: h_{t-1} = hseq[:, t-1, :] -> LDS (2048 x 16B, 8 per thread)
#pragma unroll
      for (int j = 0; j < 8; ++j) {
        int idx = tid * 8 + j;
        int b   = idx >> 6;
        int qc  = idx & 63;
        async_load_lds_b128(
            ldsa_h + (unsigned)(idx * 16),
            hseq + ((size_t)b * Tc + (t - 1)) * Hc + qc * 8);
      }
    }
    // async: this step's xg gate slice -> LDS (fire now, wait after GEMM)
    async_load_lds_b128(
        xg_dst,
        xg + ((size_t)xb * Tc + t) * G4c + xgate * Hc + hs + xhalf);
    if (t > 0) wait_asynccnt1();  // h loads done; xg may still be in flight
    __syncthreads();

    // gates(slice) = h_{t-1} @ Whh_slice^T : 16 WMMAs.
    // 4-deep A-fragment pipeline: the buffer consumed by wmma(kt) is
    // refilled for kt+4, so DS latency overlaps 3-4 WMMAs and the WAR
    // window on WMMA A-operands is comfortably cleared.
    auto lda = [&](int kt, HVec& d) {
      d.q[0] = *(const uint4*)(hrow + kt * 32 + abase);
      d.q[1] = *(const uint4*)(hrow + kt * 32 + abase + 16);
    };
    HVec abuf[4];
#pragma unroll
    for (int kt = 0; kt < 4; ++kt) lda(kt, abuf[kt]);
    v8f acc = {};
#pragma unroll
    for (int kt = 0; kt < 16; ++kt) {
      acc = __builtin_amdgcn_wmma_f32_16x16x32_f16(false, abuf[kt & 3].v,
                                                   false, bfrag[kt], (short)0,
                                                   acc, false, false);
      __builtin_amdgcn_sched_group_barrier(0x008, 1, 0);  // 1 WMMA
      if (kt + 4 < 16) {
        lda(kt + 4, abuf[kt & 3]);
        __builtin_amdgcn_sched_group_barrier(0x100, 2, 0);  // 2 DS reads
      }
    }
#pragma unroll
    for (int r = 0; r < 8; ++r)
      sh_gate[gate][mtile * 16 + abase + r][ln] = acc[r];
    wait_asynccnt0();   // xg slice now resident in LDS
    __syncthreads();

    // cell update: 512 (b,u) pairs over 256 threads
#pragma unroll
    for (int p = 0; p < 2; ++p) {
      int pi = tid + p * 256;
      int u  = pi & 15;
      int b  = pi >> 4;
      float gi = sh_gate[0][b][u] + (float)sh_xg[0][b][u];
      float gf = sh_gate[1][b][u] + (float)sh_xg[1][b][u];
      float gg = sh_gate[2][b][u] + (float)sh_xg[2][b][u];
      float go = sh_gate[3][b][u] + (float)sh_xg[3][b][u];
      float iv = sigmoidf_(gi);
      float fv = sigmoidf_(gf);
      float gv = tanhf(gg);
      float ov = sigmoidf_(go);
      float c  = fv * sh_c[b][u] + iv * gv;
      sh_c[b][u] = c;
      float h = ov * tanhf(c);
      hseq[((size_t)b * Tc + t) * Hc + hs + u] = (_Float16)h;
    }

    // grid barrier (monotonic counter, agent scope)
    __threadfence();                // release: h stores visible device-wide
    __syncthreads();
    if (tid == 0) {
      __hip_atomic_fetch_add(barrier, 1u, __ATOMIC_RELEASE,
                             __HIP_MEMORY_SCOPE_AGENT);
      unsigned target = (phase_base + (unsigned)t + 1u) * (unsigned)NWG;
      while (__hip_atomic_load(barrier, __ATOMIC_ACQUIRE,
                               __HIP_MEMORY_SCOPE_AGENT) < target) {
        __builtin_amdgcn_s_sleep(2);
      }
    }
    __syncthreads();
    __threadfence();                // acquire: see peers' h stores
  }
}

// ---------------------------------------------------------------------------
// Host launch
// ---------------------------------------------------------------------------
extern "C" void kernel_launch(void* const* d_in, const int* in_sizes, int n_in,
                              void* d_out, int out_size, void* d_ws,
                              size_t ws_size, hipStream_t stream) {
  (void)in_sizes; (void)n_in; (void)out_size; (void)ws_size;

  const float* x    = (const float*)d_in[0];
  const float* Wih0 = (const float*)d_in[1];
  const float* Whh0 = (const float*)d_in[2];
  const float* bih0 = (const float*)d_in[3];
  const float* bhh0 = (const float*)d_in[4];
  const float* Wih1 = (const float*)d_in[5];
  const float* Whh1 = (const float*)d_in[6];
  const float* bih1 = (const float*)d_in[7];
  const float* bhh1 = (const float*)d_in[8];
  const float* fcW  = (const float*)d_in[9];
  const float* fcb  = (const float*)d_in[10];
  const float* fc2W = (const float*)d_in[11];
  const float* fc2b = (const float*)d_in[12];
  float* out = (float*)d_out;

  char* ws = (char*)d_ws;
  size_t off = 0;
  auto alloc = [&](size_t bytes) -> char* {
    char* p = ws + off;
    off += (bytes + 255) & ~(size_t)255;
    return p;
  };

  unsigned*  barrier = (unsigned*)alloc(256);
  _Float16*  xg      = (_Float16*)alloc((size_t)Mc * G4c * 2);  // 128 MiB
  _Float16*  h1seq   = (_Float16*)alloc((size_t)Mc * Hc * 2);   // 32 MiB
  _Float16*  h2seq   = (_Float16*)alloc((size_t)Mc * Hc * 2);   // 32 MiB
  _Float16*  fc1buf  = (_Float16*)alloc((size_t)Mc * Oc * 2);   // 8 MiB
  _Float16*  x16     = (_Float16*)alloc((size_t)Mc * Ic * 2);
  _Float16*  wih0_16 = (_Float16*)alloc((size_t)G4c * Ic * 2);
  _Float16*  whh0_16 = (_Float16*)alloc((size_t)G4c * Hc * 2);
  _Float16*  wih1_16 = (_Float16*)alloc((size_t)G4c * Hc * 2);
  _Float16*  whh1_16 = (_Float16*)alloc((size_t)G4c * Hc * 2);
  _Float16*  fcw16   = (_Float16*)alloc((size_t)Oc * Hc * 2);
  _Float16*  fc2w16  = (_Float16*)alloc((size_t)Oc * Oc * 2);
  float*     bsum0   = (float*)alloc((size_t)G4c * 4);
  float*     bsum1   = (float*)alloc((size_t)G4c * 4);

  hipMemsetAsync(barrier, 0, 256, stream);

  auto cvt = [&](const float* src, _Float16* dst, int n) {
    f32_to_f16_kernel<<<(n + 255) / 256, 256, 0, stream>>>(src, dst, n);
  };
  cvt(x,    x16,     Mc * Ic);
  cvt(Wih0, wih0_16, G4c * Ic);
  cvt(Whh0, whh0_16, G4c * Hc);
  cvt(Wih1, wih1_16, G4c * Hc);
  cvt(Whh1, whh1_16, G4c * Hc);
  cvt(fcW,  fcw16,   Oc * Hc);
  cvt(fc2W, fc2w16,  Oc * Oc);
  bias_sum_kernel<<<(G4c + 255) / 256, 256, 0, stream>>>(bih0, bhh0, bsum0, G4c);
  bias_sum_kernel<<<(G4c + 255) / 256, 256, 0, stream>>>(bih1, bhh1, bsum1, G4c);

  // Layer 0: xg0 = x @ Wih0^T + (bih0+bhh0)   [32768 x 2048, K=128]
  wmma_gemm_kernel<_Float16><<<dim3(G4c / 64, Mc / 64), 256, 0, stream>>>(
      x16, wih0_16, bsum0, xg, Mc, G4c, Ic);
  lstm_recurrent_kernel<<<NWG, 256, 0, stream>>>(xg, whh0_16, h1seq, barrier, 0u);

  // Layer 1: xg1 = h1 @ Wih1^T + (bih1+bhh1)  [32768 x 2048, K=512]
  wmma_gemm_kernel<_Float16><<<dim3(G4c / 64, Mc / 64), 256, 0, stream>>>(
      h1seq, wih1_16, bsum1, xg, Mc, G4c, Hc);
  lstm_recurrent_kernel<<<NWG, 256, 0, stream>>>(xg, whh1_16, h2seq, barrier,
                                                 (unsigned)Tc);

  // FC chain: out = (h2 @ fcW^T + fcb) @ fc2W^T + fc2b
  wmma_gemm_kernel<_Float16><<<dim3(Oc / 64, Mc / 64), 256, 0, stream>>>(
      h2seq, fcw16, fcb, fc1buf, Mc, Oc, Hc);
  wmma_gemm_kernel<float><<<dim3(Oc / 64, Mc / 64), 256, 0, stream>>>(
      fc1buf, fc2w16, fc2b, out, Mc, Oc, Oc);
}
